// CrossAttention_20856361189385
// MI455X (gfx1250) — matrix operand order repaired and checked by reading, not compiled, not run on hardware
//
#include <hip/hip_runtime.h>

typedef _Float16 h16;
typedef __attribute__((ext_vector_type(16))) _Float16 v16h;
typedef __attribute__((ext_vector_type(8)))  _Float16 v8h;
typedef __attribute__((ext_vector_type(8)))  float    v8f;
typedef unsigned int u32;
typedef unsigned long long u64;
typedef __attribute__((ext_vector_type(4))) u32 u32x4;
typedef __attribute__((ext_vector_type(8))) int i32x8;

#define WMMA_F16(a, b, c) \
  __builtin_amdgcn_wmma_f32_16x16x32_f16(false, (a), false, (b), (short)0, (c), false, false)

static __device__ __forceinline__ v16h cat8(v8h a, v8h b) {
  return __builtin_shufflevector(a, b, 0, 1, 2, 3, 4, 5, 6, 7, 8, 9, 10, 11, 12, 13, 14, 15);
}

// ---- Tensor Data Mover helpers (D# per CDNA5 ISA 8.3/8.4) -----------------
static __device__ __forceinline__ u32x4 tdm_group0(u32 lds, u64 ga) {
  u32x4 g;
  g[0] = 1u;                                             // count=1, user mode
  g[1] = lds;                                            // lds_addr (bytes)
  g[2] = (u32)ga;                                        // global_addr[31:0]
  g[3] = (u32)((ga >> 32) & 0x01ffffffu) | (2u << 30);   // addr[56:32], type=2
  return g;
}
// 2-D tile: dim0/stride0 in elements (data_size=2B), tensor_dim1=1M,
// LDS padding: pad_interval code iv (2<<iv DWORDs), pad_amount code am (am+1 DWORDs)
static __device__ __forceinline__ i32x8 tdm_group1(u32 dim0, u32 stride0,
                                                   u32 tile0, u32 tile1,
                                                   u32 iv, u32 am) {
  i32x8 g;
  g[0] = (int)((1u << 16) | (1u << 20) | (iv << 22) | (am << 25)); // data_size=1(2B), pad_enable
  g[1] = (int)((dim0 & 0xffffu) << 16);   // tensor_dim0[15:0] @ 63:48
  g[2] = (int)(dim0 >> 16);               // tensor_dim0[31:16]; tensor_dim1[15:0]=0
  g[3] = (int)(0x10u | (tile0 << 16));    // tensor_dim1[31:16]=0x10 (1M), tile_dim0
  g[4] = (int)tile1;                      // tile_dim1; tile_dim2=0
  g[5] = (int)stride0;                    // tensor_dim0_stride[31:0]
  g[6] = 0;
  g[7] = 0;
  return g;
}
static __device__ __forceinline__ void tdm_load(u32x4 g0, i32x8 g1) {
  asm volatile("tensor_load_to_lds %0, %1" :: "s"(g0), "s"(g1) : "memory");
}

static constexpr int BATCH = 4;
static constexpr int SQ = 1024;
static constexpr int SK = 1024;
static constexpr int DMODEL = 768;
static constexpr int NH = 12;
static constexpr int HD = 64;

// ---------------------------------------------------------------------------
// RMSNorm + cast to f16: one block per row of D=768
// ---------------------------------------------------------------------------
__global__ __launch_bounds__(256) void rmsnorm_cast_kernel(
    const float* __restrict__ x, const float* __restrict__ w,
    h16* __restrict__ y, int D) {
  int row = blockIdx.x;
  const float* xr = x + (size_t)row * D;
  float ss = 0.f;
  for (int c = threadIdx.x; c < D; c += 256) {
    float v = xr[c];
    ss += v * v;
  }
#pragma unroll
  for (int off = 16; off > 0; off >>= 1) ss += __shfl_xor(ss, off, 32);
  __shared__ float sred[8];
  __shared__ float stot;
  if ((threadIdx.x & 31) == 0) sred[threadIdx.x >> 5] = ss;
  __syncthreads();
  if (threadIdx.x == 0) {
    float t = 0.f;
#pragma unroll
    for (int i = 0; i < 8; ++i) t += sred[i];
    stot = rsqrtf(t / (float)D + 1e-5f);
  }
  __syncthreads();
  float rs = stot;
  h16* yr = y + (size_t)row * D;
  for (int c = threadIdx.x; c < D; c += 256) yr[c] = (h16)(xr[c] * rs * w[c]);
}

// ---------------------------------------------------------------------------
__global__ __launch_bounds__(256) void cast_f16_kernel(
    const float* __restrict__ x, h16* __restrict__ y, size_t n) {
  size_t i = (size_t)blockIdx.x * 256 + threadIdx.x;
  if (i < n) y[i] = (h16)x[i];
}

// ---------------------------------------------------------------------------
// WMMA GEMM with TDM staging: C[M][N] f32 = A[M][K] @ Bw[N][K]^T + bias.
// Bw kept in original torch layout [N][K] so both tiles are identical-shape
// row-major 2-D TDM loads (one shared D# group1). Block tile 128x128, BK=32,
// 8 waves (4Mx2N), 32x64 per wave: 2 A-frags x 4 shared B-frags = 8 WMMA per
// K-step per wave. Double-buffered tensor_load_to_lds issued by wave 0.
// ---------------------------------------------------------------------------
__global__ __launch_bounds__(256) void gemm_f16_kernel(
    const h16* __restrict__ A, const h16* __restrict__ Bw,
    const float* __restrict__ bias, float* __restrict__ C,
    int M, int N, int K) {
  __shared__ alignas(16) h16 As[2][128][40];    // [m][k], pitch 80B
  __shared__ alignas(16) h16 Bst[2][128][40];   // [n][k], pitch 80B

  int tid = threadIdx.x;
  int wave = tid >> 5, lane = tid & 31;
  int bm = blockIdx.x * 128;
  int bn = blockIdx.y * 128;
  int wm = (wave >> 1) * 32;
  int wn = (wave & 1) * 64;

  v8f acc0[4] = {v8f{}, v8f{}, v8f{}, v8f{}};
  v8f acc1[4] = {v8f{}, v8f{}, v8f{}, v8f{}};

  int l15 = lane & 15;
  int fam = wm + l15;
  int fkg = (lane >> 4) * 16;   // A K-group
  int fbn = wn + l15;
  int fkb = (lane >> 4) * 16;   // B K-base

  u64 gaA = (u64)(uintptr_t)(A + (size_t)bm * K);
  u64 gaB = (u64)(uintptr_t)(Bw + (size_t)bn * K);
  u32 ldsA[2] = {(u32)(uintptr_t)&As[0][0][0], (u32)(uintptr_t)&As[1][0][0]};
  u32 ldsB[2] = {(u32)(uintptr_t)&Bst[0][0][0], (u32)(uintptr_t)&Bst[1][0][0]};
  // rows of 32 elems = 64B = 16 DW -> interval code 3; pad 16B = 4 DW -> code 3
  i32x8 d1 = tdm_group1((u32)K, (u32)K, 32, 128, 3, 3);

  int nk = K >> 5;
  if (tid < 32) {
    tdm_load(tdm_group0(ldsA[0], gaA), d1);
    tdm_load(tdm_group0(ldsB[0], gaB), d1);
  }
  for (int i = 0; i < nk; ++i) {
    if (tid < 32) {
      if (i + 1 < nk) {
        tdm_load(tdm_group0(ldsA[(i + 1) & 1], gaA + (size_t)(i + 1) * 64), d1);
        tdm_load(tdm_group0(ldsB[(i + 1) & 1], gaB + (size_t)(i + 1) * 64), d1);
        __builtin_amdgcn_s_wait_tensorcnt(2);  // tile i complete (in-order)
      } else {
        __builtin_amdgcn_s_wait_tensorcnt(0);
      }
    }
    __syncthreads();  // publish tile i
    int bi = i & 1;
    v16h af0 = cat8(*(const v8h*)&As[bi][fam][fkg], *(const v8h*)&As[bi][fam][fkg + 8]);
    v16h af1 = cat8(*(const v8h*)&As[bi][fam + 16][fkg], *(const v8h*)&As[bi][fam + 16][fkg + 8]);
#pragma unroll
    for (int t = 0; t < 4; ++t) {
      int n0 = fbn + t * 16;
      v16h bf = cat8(*(const v8h*)&Bst[bi][n0][fkb], *(const v8h*)&Bst[bi][n0][fkb + 8]);
      acc0[t] = WMMA_F16(af0, bf, acc0[t]);
      acc1[t] = WMMA_F16(af1, bf, acc1[t]);
    }
    __syncthreads();  // all waves done with tile i before its buffer is reissued
  }

  int rg = (lane >> 4) * 8;
#pragma unroll
  for (int r = 0; r < 8; ++r) {
    int m0 = bm + wm + rg + r;
#pragma unroll
    for (int t = 0; t < 4; ++t) {
      int n0 = bn + wn + t * 16 + l15;
      float bv = bias[n0];
      C[(size_t)m0 * N + n0]        = acc0[t][r] + bv;
      C[(size_t)(m0 + 16) * N + n0] = acc1[t][r] + bv;
    }
  }
}

// ---------------------------------------------------------------------------
// ND mixed RoPE + cast to f16, head-major output [B,H,S,64]
// ---------------------------------------------------------------------------
__global__ __launch_bounds__(256) void rope_cast_kernel(
    const float* __restrict__ X, const int* __restrict__ pos,
    const float* __restrict__ freqs, h16* __restrict__ Out,
    int S, int ldx) {
  int idx = blockIdx.x * 256 + threadIdx.x;
  int total = BATCH * S * NH * 32;
  if (idx >= total) return;
  int f = idx & 31;
  int t = idx >> 5;
  int h = t % NH; t /= NH;
  int s = t % S;
  int b = t / S;
  float p0 = (float)pos[((size_t)b * S + s) * 2 + 0];
  float p1 = (float)pos[((size_t)b * S + s) * 2 + 1];
  float ang = p0 * freqs[h * 32 + f] + p1 * freqs[NH * 32 + h * 32 + f];
  float cn = cosf(ang), sn = sinf(ang);
  const float* xr = X + ((size_t)b * S + s) * ldx + h * 64 + 2 * f;
  float xre = xr[0], xim = xr[1];
  h16* o = Out + (((size_t)(b * NH + h)) * S + s) * 64 + 2 * f;
  o[0] = (h16)(xre * cn - xim * sn);
  o[1] = (h16)(xre * sn + xim * cn);
}

// Repack V TRANSPOSED: kvp[B,S,1536] cols 768.. -> f16 [B,H,64,S]
// (d-major so attention V tiles are plain 2-D TDM loads; TDM cannot transpose)
__global__ __launch_bounds__(256) void v_cast_kernel(
    const float* __restrict__ kvp, h16* __restrict__ Vout, int S) {
  size_t idx = (size_t)blockIdx.x * 256 + threadIdx.x;
  size_t total = (size_t)BATCH * NH * 64 * S;
  if (idx >= total) return;
  int s = (int)(idx % S);
  size_t t = idx / S;
  int d = (int)(t % 64); t /= 64;
  int h = (int)(t % NH);
  int b = (int)(t / NH);
  Vout[idx] = (h16)kvp[((size_t)b * S + s) * (2 * DMODEL) + DMODEL + h * 64 + d];
}

// ---------------------------------------------------------------------------
// Flash attention with TDM-staged K/V tiles, double buffered.
// grid (Sq/128, B*H). 8 waves/block, 16 q-rows per wave, 32 k-cols per step.
// ---------------------------------------------------------------------------
__global__ __launch_bounds__(256) void attn_kernel(
    const h16* __restrict__ Q,   // [B,H,S,64]
    const h16* __restrict__ Kt,  // [B,H,S,64]
    const h16* __restrict__ V,   // [B,H,64,S]  (transposed)
    h16* __restrict__ O,         // [B,S,768]
    int S) {
  __shared__ alignas(16) h16 Ks[2][32][72];     // [k][d], pitch 144B
  __shared__ alignas(16) h16 Vt[2][64][40];     // [d][k], pitch 80B
  __shared__ alignas(16) h16 Ps[8][16][32];     // per-wave P relay

  int bh = blockIdx.y;
  int b = bh / NH, h = bh % NH;
  const h16* Qb = Q + (size_t)bh * S * 64;
  const h16* Kb = Kt + (size_t)bh * S * 64;
  const h16* Vb = V + (size_t)bh * 64 * S;

  int tid = threadIdx.x, wave = tid >> 5, lane = tid & 31;
  int qbase = blockIdx.x * 128 + wave * 16;

  int l15 = lane & 15;
  int kg = (lane >> 4) * 8;
  int kb = (lane >> 4) * 16;

  v16h aq0, aq1;
  {
    const h16* qrow = Qb + (size_t)(qbase + l15) * 64;
    aq0 = cat8(*(const v8h*)(qrow + kb), *(const v8h*)(qrow + kb + 8));
    aq1 = cat8(*(const v8h*)(qrow + 32 + kb), *(const v8h*)(qrow + 40 + kb));
  }

  const float scale = 0.125f;
  float mrow[8], lrow[8];
#pragma unroll
  for (int r = 0; r < 8; ++r) { mrow[r] = -1e30f; lrow[r] = 0.f; }
  v8f o0 = {}, o1 = {}, o2 = {}, o3 = {};

  u64 gaK = (u64)(uintptr_t)Kb;
  u64 gaV = (u64)(uintptr_t)Vb;
  u32 ldsK[2] = {(u32)(uintptr_t)&Ks[0][0][0], (u32)(uintptr_t)&Ks[1][0][0]};
  u32 ldsV[2] = {(u32)(uintptr_t)&Vt[0][0][0], (u32)(uintptr_t)&Vt[1][0][0]};
  // K tile: rows of 64 elems = 128B = 32 DW -> interval code 4; pad 4 DW -> 3
  i32x8 dK = tdm_group1(64, 64, 64, 32, 4, 3);
  // V tile: rows of 32 elems = 64B = 16 DW -> interval code 3; pad 4 DW -> 3
  i32x8 dV = tdm_group1((u32)S, (u32)S, 32, 64, 3, 3);

  int nk = S >> 5;
  if (tid < 32) {
    tdm_load(tdm_group0(ldsK[0], gaK), dK);
    tdm_load(tdm_group0(ldsV[0], gaV), dV);
  }
  for (int i = 0; i < nk; ++i) {
    if (tid < 32) {
      if (i + 1 < nk) {
        tdm_load(tdm_group0(ldsK[(i + 1) & 1], gaK + (size_t)(i + 1) * 4096), dK);
        tdm_load(tdm_group0(ldsV[(i + 1) & 1], gaV + (size_t)(i + 1) * 64), dV);
        __builtin_amdgcn_s_wait_tensorcnt(2);
      } else {
        __builtin_amdgcn_s_wait_tensorcnt(0);
      }
    }
    __syncthreads();
    int bi = i & 1;

    // scores
    v8f s0 = {}, s1 = {};
    {
      v16h bk0 = cat8(*(const v8h*)&Ks[bi][l15][kb], *(const v8h*)&Ks[bi][l15][kb + 8]);
      v16h bk1 = cat8(*(const v8h*)&Ks[bi][l15][32 + kb], *(const v8h*)&Ks[bi][l15][40 + kb]);
      v16h bk2 = cat8(*(const v8h*)&Ks[bi][l15 + 16][kb], *(const v8h*)&Ks[bi][l15 + 16][kb + 8]);
      v16h bk3 = cat8(*(const v8h*)&Ks[bi][l15 + 16][32 + kb], *(const v8h*)&Ks[bi][l15 + 16][40 + kb]);
      s0 = WMMA_F16(aq0, bk0, s0);
      s0 = WMMA_F16(aq1, bk1, s0);
      s1 = WMMA_F16(aq0, bk2, s1);
      s1 = WMMA_F16(aq1, bk3, s1);
    }

    // online softmax
    float tmax[8], corr[8], esum[8];
#pragma unroll
    for (int r = 0; r < 8; ++r) {
      s0[r] *= scale;
      s1[r] *= scale;
      tmax[r] = fmaxf(s0[r], s1[r]);
    }
#pragma unroll
    for (int off = 1; off < 16; off <<= 1) {
#pragma unroll
      for (int r = 0; r < 8; ++r)
        tmax[r] = fmaxf(tmax[r], __shfl_xor(tmax[r], off, 32));
    }
#pragma unroll
    for (int r = 0; r < 8; ++r) {
      float nm = fmaxf(mrow[r], tmax[r]);
      corr[r] = __expf(mrow[r] - nm);
      mrow[r] = nm;
      s0[r] = __expf(s0[r] - nm);
      s1[r] = __expf(s1[r] - nm);
      esum[r] = s0[r] + s1[r];
    }
#pragma unroll
    for (int off = 1; off < 16; off <<= 1) {
#pragma unroll
      for (int r = 0; r < 8; ++r) esum[r] += __shfl_xor(esum[r], off, 32);
    }
#pragma unroll
    for (int r = 0; r < 8; ++r) {
      lrow[r] = lrow[r] * corr[r] + esum[r];
      o0[r] *= corr[r];
      o1[r] *= corr[r];
      o2[r] *= corr[r];
      o3[r] *= corr[r];
    }

    // P: C layout -> wave-private LDS -> A layout
    {
      int rg = (lane >> 4) * 8;
#pragma unroll
      for (int r = 0; r < 8; ++r) {
        Ps[wave][rg + r][l15]      = (h16)s0[r];
        Ps[wave][rg + r][l15 + 16] = (h16)s1[r];
      }
    }
    v16h ap = cat8(*(const v8h*)&Ps[wave][l15][kb], *(const v8h*)&Ps[wave][l15][kb + 8]);

    // O += P @ V
    v16h bv0 = cat8(*(const v8h*)&Vt[bi][l15][kb], *(const v8h*)&Vt[bi][l15][kb + 8]);
    v16h bv1 = cat8(*(const v8h*)&Vt[bi][l15 + 16][kb], *(const v8h*)&Vt[bi][l15 + 16][kb + 8]);
    v16h bv2 = cat8(*(const v8h*)&Vt[bi][l15 + 32][kb], *(const v8h*)&Vt[bi][l15 + 32][kb + 8]);
    v16h bv3 = cat8(*(const v8h*)&Vt[bi][l15 + 48][kb], *(const v8h*)&Vt[bi][l15 + 48][kb + 8]);
    o0 = WMMA_F16(ap, bv0, o0);
    o1 = WMMA_F16(ap, bv1, o1);
    o2 = WMMA_F16(ap, bv2, o2);
    o3 = WMMA_F16(ap, bv3, o3);

    __syncthreads();  // done reading tile i before its buffer is reissued
  }

  int rg = (lane >> 4) * 8;
#pragma unroll
  for (int r = 0; r < 8; ++r) {
    float inv = 1.0f / lrow[r];
    int qrow = qbase + rg + r;
    size_t base = ((size_t)b * S + qrow) * DMODEL + h * 64;
    O[base + l15]      = (h16)(o0[r] * inv);
    O[base + l15 + 16] = (h16)(o1[r] * inv);
    O[base + l15 + 32] = (h16)(o2[r] * inv);
    O[base + l15 + 48] = (h16)(o3[r] * inv);
  }
}

// ---------------------------------------------------------------------------
extern "C" void kernel_launch(void* const* d_in, const int* in_sizes, int n_in,
                              void* d_out, int out_size, void* d_ws, size_t ws_size,
                              hipStream_t stream) {
  const float* q      = (const float*)d_in[0];
  const float* kv     = (const float*)d_in[1];
  const int*   posq   = (const int*)d_in[2];
  const int*   posk   = (const int*)d_in[3];
  const float* w_norm = (const float*)d_in[4];
  const float* w_q    = (const float*)d_in[5];
  const float* b_q    = (const float*)d_in[6];
  const float* w_kv   = (const float*)d_in[7];
  const float* b_kv   = (const float*)d_in[8];
  const float* w_out  = (const float*)d_in[9];
  const float* b_out  = (const float*)d_in[10];
  const float* freqs  = (const float*)d_in[11];
  float* out = (float*)d_out;

  const int M = BATCH * SQ;  // 4096

  char* w = (char*)d_ws;
  auto alloc = [&](size_t bytes) -> char* {
    char* p = w;
    w += (bytes + 255) & ~(size_t)255;
    return p;
  };
  h16*   qn16   = (h16*)alloc((size_t)M * DMODEL * 2);
  h16*   kv16   = (h16*)alloc((size_t)M * DMODEL * 2);
  h16*   wq16   = (h16*)alloc((size_t)DMODEL * DMODEL * 2);     // [N][K] torch layout
  h16*   wkv16  = (h16*)alloc((size_t)2 * DMODEL * DMODEL * 2); // [2D][D]
  h16*   wout16 = (h16*)alloc((size_t)DMODEL * DMODEL * 2);
  float* qp     = (float*)alloc((size_t)M * DMODEL * 4);
  float* kvp    = (float*)alloc((size_t)M * 2 * DMODEL * 4);
  h16*   qr16   = (h16*)alloc((size_t)BATCH * NH * SQ * HD * 2);
  h16*   kr16   = (h16*)alloc((size_t)BATCH * NH * SK * HD * 2);
  h16*   v16    = (h16*)alloc((size_t)BATCH * NH * HD * SK * 2);
  h16*   o16    = (h16*)alloc((size_t)M * DMODEL * 2);
  (void)ws_size; (void)n_in; (void)in_sizes; (void)out_size;

  rmsnorm_cast_kernel<<<M, 256, 0, stream>>>(q, w_norm, qn16, DMODEL);

  {
    size_t n = (size_t)M * DMODEL;
    cast_f16_kernel<<<(unsigned)((n + 255) / 256), 256, 0, stream>>>(kv, kv16, n);
  }
  {
    size_t n = (size_t)DMODEL * DMODEL;
    cast_f16_kernel<<<(unsigned)((n + 255) / 256), 256, 0, stream>>>(w_q, wq16, n);
    cast_f16_kernel<<<(unsigned)((n + 255) / 256), 256, 0, stream>>>(w_out, wout16, n);
    size_t n2 = (size_t)2 * DMODEL * DMODEL;
    cast_f16_kernel<<<(unsigned)((n2 + 255) / 256), 256, 0, stream>>>(w_kv, wkv16, n2);
  }

  {
    dim3 g(M / 128, DMODEL / 128);
    gemm_f16_kernel<<<g, 256, 0, stream>>>(qn16, wq16, b_q, qp, M, DMODEL, DMODEL);
  }
  {
    dim3 g(M / 128, 2 * DMODEL / 128);
    gemm_f16_kernel<<<g, 256, 0, stream>>>(kv16, wkv16, b_kv, kvp, M, 2 * DMODEL, DMODEL);
  }

  {
    int total = BATCH * SQ * NH * 32;
    rope_cast_kernel<<<(total + 255) / 256, 256, 0, stream>>>(qp, posq, freqs, qr16, SQ, DMODEL);
    rope_cast_kernel<<<(total + 255) / 256, 256, 0, stream>>>(kvp, posk, freqs, kr16, SK, 2 * DMODEL);
    size_t nv = (size_t)BATCH * NH * 64 * SK;
    v_cast_kernel<<<(unsigned)((nv + 255) / 256), 256, 0, stream>>>(kvp, v16, SK);
  }

  {
    dim3 g(SQ / 128, BATCH * NH);
    attn_kernel<<<g, 256, 0, stream>>>(qr16, kr16, v16, o16, SK);
  }

  {
    dim3 g(M / 128, DMODEL / 128);
    gemm_f16_kernel<<<g, 256, 0, stream>>>(o16, wout16, b_out, out, M, DMODEL, DMODEL);
  }
}
